// ALiBiMultiHeadAttention_35029753266483
// MI455X (gfx1250) — compile-verified, matrix operand order checked by reading, not run on hardware
//
#include <hip/hip_runtime.h>
#include <hip/hip_bf16.h>
#include <math.h>

// ALiBi multi-head attention, CDNA5 (gfx1250), wave32, bf16 WMMA w/ fp32 accum.
// B=2, S=2048, E=1024, H=16, D=64.

typedef __attribute__((ext_vector_type(16))) __bf16 v16bf;
typedef __attribute__((ext_vector_type(8)))  float  v8f;
typedef __attribute__((ext_vector_type(4)))  int    v4i;

#define E_DIM 1024
#define S_LEN 2048
#define NH    16
#define HD    64

#if __has_builtin(__builtin_amdgcn_global_load_async_to_lds_b128)
#define HAVE_ASYNC 1
#else
#define HAVE_ASYNC 0
#endif

// 16-byte global->LDS copy; async LDS-DMA (ASYNCcnt) when available.
// Builtin signature (from hipcc diagnostic): arg0 = v4i addrspace(1)* (global
// src), arg1 = v4i addrspace(3)* (LDS dst), then imm offset + cpol.
__device__ __forceinline__ void cp_b128(void* lds, const void* g) {
#if HAVE_ASYNC
  __builtin_amdgcn_global_load_async_to_lds_b128(
      (__attribute__((address_space(1))) v4i*)g,
      (__attribute__((address_space(3))) v4i*)lds, 0, 0);
#else
  *(uint4*)lds = *(const uint4*)g;
#endif
}

__device__ __forceinline__ void wait_async() {
#if HAVE_ASYNC
  asm volatile("s_wait_asynccnt 0" ::: "memory");
#endif
}

// Pack two f32 into packed bf16 (compiler emits v_cvt_pk_bf16_f32).
__device__ __forceinline__ unsigned pk2(float a, float b) {
  union { __bf16 h[2]; unsigned u; } p;
  p.h[0] = (__bf16)a; p.h[1] = (__bf16)b;
  return p.u;
}

// ---- CDNA5 WMMA 16x16x32 bf16 fragment helpers -----------------------------
// A-matrix (16x32 bf16): lane L -> M = L%16, half h = L/16,
// VGPR pair v (0..7): K = 16*(v/4) + 2*(v%4) + 8*h + {0,1}.
// B-matrix (32x16) mirrored: lane L -> N = L%16, same K striping.
// C/D (16x16 f32): VGPR r -> M = r + 8*h, N = L%16.
__device__ __forceinline__ v16bf load_frag16(const __bf16* tile, int ld, int row, int h) {
  v16bf f;
#pragma unroll
  for (int i = 0; i < 16; ++i) {
    int v = i >> 1;
    int k = 16 * (v >> 2) + 2 * (v & 3) + 8 * h + (i & 1);
    f[i] = tile[row * ld + k];
  }
  return f;
}

__device__ __forceinline__ v8f wmma_bf16(v16bf a, v16bf b, v8f c) {
  return __builtin_amdgcn_wmma_f32_16x16x32_bf16(
      /*neg_a=*/false, a, /*neg_b=*/false, b,
      /*c_mod=*/(short)0, c, /*reuse_a=*/false, /*reuse_b=*/false);
}

// ---- Dense GEMM: A[M,K] @ B[K,N] + bias[N] (fp32 in, bf16 compute) ---------
// OUT_MODE 0: fp32 C[M,N]           (final O projection)
// OUT_MODE 1: bf16 heads [B,H,S,D]  (Q, K)
// OUT_MODE 2: bf16 heads [B,H,D,S]  (V, pre-transposed for attention)
// Workgroup = 256 thr = 8 waves as 4(M) x 2(N); tile 128x128, K-step 64.
template <int OUT_MODE>
__global__ __launch_bounds__(256)
void gemm_bias(const float* __restrict__ A, const float* __restrict__ B,
               const float* __restrict__ bias, float* __restrict__ Cf,
               __bf16* __restrict__ Cb, int M, int N, int K) {
  __shared__ __bf16 As[128][72];  // row-major A tile (rows 144B: 8/16B aligned)
  __shared__ __bf16 Bs[128][72];  // B tile transposed: Bs[n][k], k-pairs packed

  const int tid  = threadIdx.x;
  const int lane = tid & 31;
  const int wid  = tid >> 5;
  const int h    = lane >> 4;
  const int ln   = lane & 15;
  const int wm   = wid & 3;   // 32-row slice
  const int wn   = wid >> 2;  // 64-col slice
  const int m0 = blockIdx.y * 128;
  const int n0 = blockIdx.x * 128;

  v8f acc[2][4];
#pragma unroll
  for (int i = 0; i < 2; ++i)
#pragma unroll
    for (int j = 0; j < 4; ++j)
#pragma unroll
      for (int r = 0; r < 8; ++r) acc[i][j][r] = 0.0f;

  for (int k0 = 0; k0 < K; k0 += 64) {
    __syncthreads();
    // A tile 128x64: float4 loads, packed-bf16 uint2 stores.
#pragma unroll
    for (int e = tid; e < 128 * 16; e += 256) {
      int r = e >> 4, c4 = (e & 15) << 2;
      float4 a4 = *(const float4*)(A + (size_t)(m0 + r) * K + k0 + c4);
      uint2 t;
      t.x = pk2(a4.x, a4.y);
      t.y = pk2(a4.z, a4.w);
      *(uint2*)&As[r][c4] = t;
    }
    // B tile 64x128 -> Bs[n][k]: read two k-rows (coalesced over n),
    // store each k-pair as one packed dword so fragment pairs stay contiguous.
#pragma unroll
    for (int e = tid; e < 32 * 128; e += 256) {
      int kp = e >> 7, n = e & 127;
      float b0 = B[(size_t)(k0 + 2 * kp) * N + n0 + n];
      float b1 = B[(size_t)(k0 + 2 * kp + 1) * N + n0 + n];
      *(unsigned*)&Bs[n][2 * kp] = pk2(b0, b1);
    }
    __syncthreads();

#pragma unroll
    for (int kh = 0; kh < 2; ++kh) {
      v16bf af[2], bf[4];
#pragma unroll
      for (int i = 0; i < 2; ++i)
        af[i] = load_frag16(&As[wm * 32 + i * 16][kh * 32], 72, ln, h);
#pragma unroll
      for (int j = 0; j < 4; ++j)
        bf[j] = load_frag16(&Bs[wn * 64 + j * 16][kh * 32], 72, ln, h);
#pragma unroll
      for (int i = 0; i < 2; ++i)
#pragma unroll
        for (int j = 0; j < 4; ++j)
          acc[i][j] = wmma_bf16(af[i], bf[j], acc[i][j]);
    }
  }

#pragma unroll
  for (int i = 0; i < 2; ++i)
#pragma unroll
    for (int j = 0; j < 4; ++j)
#pragma unroll
      for (int r = 0; r < 8; ++r) {
        int m = m0 + wm * 32 + i * 16 + r + 8 * h;
        int n = n0 + wn * 64 + j * 16 + ln;
        float val = acc[i][j][r] + bias[n];
        if (OUT_MODE == 0) {
          Cf[(size_t)m * N + n] = val;
        } else {
          int b = m >> 11, s = m & (S_LEN - 1);
          int hh = n >> 6, d = n & (HD - 1);
          size_t idx = (OUT_MODE == 1)
              ? ((((size_t)b * NH + hh) * S_LEN + s) * HD + d)
              : ((((size_t)b * NH + hh) * HD + d) * S_LEN + s);
          Cb[idx] = (__bf16)val;
        }
      }
}

// ---- Flash-style ALiBi attention -------------------------------------------
// Q,K: bf16 [B,H,S,D]; V: bf16 [B,H,D,S]; ctx out: fp32 [B,S,E].
// grid = (S/128, H, B); block = 256 (8 waves); wave w owns 16 query rows.
// K/V tiles (32 keys) double-buffered via async LDS-DMA.
__global__ __launch_bounds__(256)
void attn_alibi(const __bf16* __restrict__ Qb, const __bf16* __restrict__ Kb,
                const __bf16* __restrict__ Vb, float* __restrict__ ctx) {
  __shared__ __bf16 Qlds[128][72];       // [row][d]   rows 144B
  __shared__ __bf16 Klds[2][32][72];     // [key][d]   rows 144B
  __shared__ __bf16 Vt[2][64][40];       // [d][key]   rows 80B
  __shared__ __bf16 Plds[8][16][34];     // per-wave P staging (C->A transpose)

  const int tid  = threadIdx.x;
  const int lane = tid & 31;
  const int wid  = tid >> 5;
  const int h    = lane >> 4;
  const int ln   = lane & 15;

  const int q0    = blockIdx.x * 128;
  const int head  = blockIdx.y;
  const int batch = blockIdx.z;

  const float scale = 0.125f;                            // 1/sqrt(64)
  const float slope = exp2f(-0.5f * (float)(head + 1));  // (2^8)^(-(h+1)/16)
  const size_t hb = ((size_t)batch * NH + head) * (size_t)S_LEN * HD;
  const __bf16* qh = Qb + hb;
  const __bf16* kh_ = Kb + hb;
  const __bf16* vh = Vb + hb;  // [D][S] within head

  // Stage Q block (128x64 bf16, contiguous) via b128 copies.
#pragma unroll
  for (int e = tid; e < 128 * 8; e += 256) {
    int r = e >> 3, c8 = (e & 7) << 3;
    cp_b128(&Qlds[r][c8], qh + (size_t)(q0 + r) * HD + c8);
  }
  // Prefetch first K/V tile into buffer 0.
  {
    int r = tid >> 3, c8 = (tid & 7) << 3;
    cp_b128(&Klds[0][r][c8], kh_ + (size_t)r * HD + c8);
    int rv = tid >> 2, cv = (tid & 3) << 3;
    cp_b128(&Vt[0][rv][cv], vh + (size_t)rv * S_LEN + cv);
  }
  wait_async();
  __syncthreads();

  v16bf qf[2];
  qf[0] = load_frag16(&Qlds[wid * 16][0], 72, ln, h);   // d 0..31
  qf[1] = load_frag16(&Qlds[wid * 16][32], 72, ln, h);  // d 32..63

  float mrow[8], lrow[8];
  v8f acc[4];
#pragma unroll
  for (int r = 0; r < 8; ++r) { mrow[r] = -1.0e30f; lrow[r] = 0.0f; }
#pragma unroll
  for (int nt = 0; nt < 4; ++nt)
#pragma unroll
    for (int r = 0; r < 8; ++r) acc[nt][r] = 0.0f;

  const int T = S_LEN / 32;
  for (int t = 0; t < T; ++t) {
    const int buf = t & 1;
    const int j0 = t * 32;

    // Kick off next tile into the other buffer (its last readers passed the
    // barrier that ended iteration t-1).
    if (t + 1 < T) {
      int j1 = j0 + 32;
      int r = tid >> 3, c8 = (tid & 7) << 3;
      cp_b128(&Klds[buf ^ 1][r][c8], kh_ + (size_t)(j1 + r) * HD + c8);
      int rv = tid >> 2, cv = (tid & 3) << 3;
      cp_b128(&Vt[buf ^ 1][rv][cv], vh + (size_t)rv * S_LEN + j1 + cv);
    }

    // S = Q.K^T for two 16-key tiles (d = 64 -> 2 WMMAs each)
    v8f s[2];
#pragma unroll
    for (int jt = 0; jt < 2; ++jt) {
      v16bf kf0 = load_frag16(&Klds[buf][jt * 16][0], 72, ln, h);
      v16bf kf1 = load_frag16(&Klds[buf][jt * 16][32], 72, ln, h);
      v8f c0;
#pragma unroll
      for (int r = 0; r < 8; ++r) c0[r] = 0.0f;
      c0 = wmma_bf16(qf[0], kf0, c0);
      c0 = wmma_bf16(qf[1], kf1, c0);
      s[jt] = c0;
    }

    // scale + ALiBi bias: element (i = q0+16w+r+8h, j = j0+16jt+ln)
#pragma unroll
    for (int jt = 0; jt < 2; ++jt)
#pragma unroll
      for (int r = 0; r < 8; ++r) {
        float i_idx = (float)(q0 + wid * 16 + r + 8 * h);
        float j_idx = (float)(j0 + jt * 16 + ln);
        s[jt][r] = s[jt][r] * scale + slope * (j_idx - i_idx);
      }

    // Online softmax (row reductions across the 16-lane N-groups)
#pragma unroll
    for (int r = 0; r < 8; ++r) {
      float tmx = fmaxf(s[0][r], s[1][r]);
#pragma unroll
      for (int msk = 8; msk >= 1; msk >>= 1) tmx = fmaxf(tmx, __shfl_xor(tmx, msk, 32));
      float mn = fmaxf(mrow[r], tmx);
      float alpha = __expf(mrow[r] - mn);
      s[0][r] = __expf(s[0][r] - mn);
      s[1][r] = __expf(s[1][r] - mn);
      float ssum = s[0][r] + s[1][r];
#pragma unroll
      for (int msk = 8; msk >= 1; msk >>= 1) ssum += __shfl_xor(ssum, msk, 32);
      lrow[r] = lrow[r] * alpha + ssum;
      mrow[r] = mn;
#pragma unroll
      for (int nt = 0; nt < 4; ++nt) acc[nt][r] *= alpha;
    }

    // Transpose P (C layout) -> A layout via per-wave LDS pad
#pragma unroll
    for (int r = 0; r < 8; ++r) {
      Plds[wid][r + 8 * h][ln]      = (__bf16)s[0][r];
      Plds[wid][r + 8 * h][16 + ln] = (__bf16)s[1][r];
    }
    asm volatile("s_wait_dscnt 0" ::: "memory");  // wave-local LDS RAW
    v16bf pf = load_frag16(&Plds[wid][0][0], 34, ln, h);

    // ctx += P (16x32) x V (32x64)
#pragma unroll
    for (int nt = 0; nt < 4; ++nt) {
      v16bf vf = load_frag16(&Vt[buf][nt * 16][0], 40, ln, h);
      acc[nt] = wmma_bf16(pf, vf, acc[nt]);
    }

    // Next tile must be fully landed and all waves done with this one.
    wait_async();
    __syncthreads();
  }

  // Normalize and store context in [B,S,E] fp32 (head slice)
#pragma unroll
  for (int nt = 0; nt < 4; ++nt)
#pragma unroll
    for (int r = 0; r < 8; ++r) {
      int row = q0 + wid * 16 + r + 8 * h;
      int col = head * HD + nt * 16 + ln;
      ctx[((size_t)batch * S_LEN + row) * E_DIM + col] = acc[nt][r] / lrow[r];
    }
}

// ---- Launcher ---------------------------------------------------------------
extern "C" void kernel_launch(void* const* d_in, const int* in_sizes, int n_in,
                              void* d_out, int out_size, void* d_ws, size_t ws_size,
                              hipStream_t stream) {
  const float* x  = (const float*)d_in[0];
  const float* Wq = (const float*)d_in[1];
  const float* bq = (const float*)d_in[2];
  const float* Wk = (const float*)d_in[3];
  const float* bk = (const float*)d_in[4];
  const float* Wv = (const float*)d_in[5];
  const float* bv = (const float*)d_in[6];
  const float* Wo = (const float*)d_in[7];
  const float* bo = (const float*)d_in[8];
  float* out = (float*)d_out;

  const int M = 2 * S_LEN;                    // 4096 rows
  const size_t elems = (size_t)M * E_DIM;     // 4M elements
  // Workspace: qbf/kbf/vbf (bf16, 8MB each) + ctx (fp32, 16MB) = 40MB.
  __bf16* qbf = (__bf16*)d_ws;
  __bf16* kbf = qbf + elems;
  __bf16* vbf = kbf + elems;
  float*  ctx = (float*)(vbf + elems);

  dim3 gemm_grid(E_DIM / 128, M / 128);  // (8, 32)
  dim3 blk(256);

  gemm_bias<1><<<gemm_grid, blk, 0, stream>>>(x, Wq, bq, nullptr, qbf, M, E_DIM, E_DIM);
  gemm_bias<1><<<gemm_grid, blk, 0, stream>>>(x, Wk, bk, nullptr, kbf, M, E_DIM, E_DIM);
  gemm_bias<2><<<gemm_grid, blk, 0, stream>>>(x, Wv, bv, nullptr, vbf, M, E_DIM, E_DIM);

  dim3 attn_grid(S_LEN / 128, NH, 2);    // (16, 16, 2)
  attn_alibi<<<attn_grid, blk, 0, stream>>>(qbf, kbf, vbf, ctx);

  gemm_bias<0><<<gemm_grid, blk, 0, stream>>>(ctx, Wo, bo, out, nullptr, M, E_DIM, E_DIM);
}
